// MyViT_7816840478742
// MI455X (gfx1250) — compile-verified, matrix operand order checked by reading, not run on hardware
//
#include <hip/hip_runtime.h>
#include <math.h>

#define L_   8
#define D_   768
#define H_   12
#define DH_  64
#define NP_  14
#define PS_  16
#define C_   3
#define OUT_ 1000
#define NB_  32
#define S_   197
#define P_   196
#define FF_  3072
#define IMGW 224
#define ATTLD 224          // padded leading dim for att rows and v^T rows (mult of 32)

typedef __attribute__((ext_vector_type(16))) _Float16 v16h;
typedef __attribute__((ext_vector_type(8)))  _Float16 v8h;
typedef __attribute__((ext_vector_type(8)))  float    v8f;

// ---------------- WMMA fragment helpers (CDNA5 wave32 layouts) ----------------
// 16-bit fragment: lane = idx(0..15) + 16*half; element e -> k = 16*(e>>3) + 8*half + (e&7)
// => per lane: two contiguous 8-half (16B) runs at k0+8*half and k0+16+8*half.

template<bool CLAMP_M>
__device__ __forceinline__ v16h load_a16(const _Float16* A, int lda, int m0, int k0, int Mlim) {
  const int lane = threadIdx.x & 31;
  int m = m0 + (lane & 15);
  const int half = lane >> 4;
  if (CLAMP_M) m = (m < Mlim) ? m : (Mlim - 1);   // always-valid load; rows >= Mlim never stored
  const _Float16* base = A + (size_t)m * lda + k0 + (half << 3);
  v8h lo = *(const v8h*)(base);
  v8h hi = *(const v8h*)(base + 16);
  v16h r;
#pragma unroll
  for (int e = 0; e < 8; ++e) { r[e] = lo[e]; r[8 + e] = hi[e]; }
  return r;
}

// B fragment from TRANSPOSED source: Bt is (N x K) row-major f16; B[k][n] = Bt[n][k].
template<bool CLAMP_N>
__device__ __forceinline__ v16h load_bt16(const _Float16* Bt, int ldb, int k0, int n0, int Nlim) {
  const int lane = threadIdx.x & 31;
  int n = n0 + (lane & 15);
  const int half = lane >> 4;
  if (CLAMP_N) n = (n < Nlim) ? n : (Nlim - 1);
  const _Float16* base = Bt + (size_t)n * ldb + k0 + (half << 3);
  v8h lo = *(const v8h*)(base);
  v8h hi = *(const v8h*)(base + 16);
  v16h r;
#pragma unroll
  for (int e = 0; e < 8; ++e) { r[e] = lo[e]; r[8 + e] = hi[e]; }
  return r;
}

__device__ __forceinline__ v8f wmma_f16(v16h a, v16h b, v8f c) {
  return __builtin_amdgcn_wmma_f32_16x16x32_f16(false, a, false, b, (short)0, c,
                                                false, false);
}

// ---- software-pipelined 16x64 tile: prefetch next k-step while WMMAs run ----
template<bool CLAMP_M, bool CLAMP_N>
__device__ __forceinline__ void gemm_16x64(const _Float16* __restrict__ A, int lda, int Mlim,
                                           const _Float16* __restrict__ Bt, int ldb, int Nlim,
                                           int m0, int n0, int K, v8f acc[4]) {
  v16h a  = load_a16<CLAMP_M>(A, lda, m0, 0, Mlim);
  v16h b0 = load_bt16<CLAMP_N>(Bt, ldb, 0, n0,      Nlim);
  v16h b1 = load_bt16<CLAMP_N>(Bt, ldb, 0, n0 + 16, Nlim);
  v16h b2 = load_bt16<CLAMP_N>(Bt, ldb, 0, n0 + 32, Nlim);
  v16h b3 = load_bt16<CLAMP_N>(Bt, ldb, 0, n0 + 48, Nlim);
  for (int k0 = 0; k0 < K; k0 += 32) {
    int k1 = k0 + 32; if (k1 >= K) k1 = 0;   // wrap => unconditional, branch-free prefetch
    v16h na  = load_a16<CLAMP_M>(A, lda, m0, k1, Mlim);
    v16h nb0 = load_bt16<CLAMP_N>(Bt, ldb, k1, n0,      Nlim);
    v16h nb1 = load_bt16<CLAMP_N>(Bt, ldb, k1, n0 + 16, Nlim);
    v16h nb2 = load_bt16<CLAMP_N>(Bt, ldb, k1, n0 + 32, Nlim);
    v16h nb3 = load_bt16<CLAMP_N>(Bt, ldb, k1, n0 + 48, Nlim);
    acc[0] = wmma_f16(a, b0, acc[0]);
    acc[1] = wmma_f16(a, b1, acc[1]);
    acc[2] = wmma_f16(a, b2, acc[2]);
    acc[3] = wmma_f16(a, b3, acc[3]);
    a = na; b0 = nb0; b1 = nb1; b2 = nb2; b3 = nb3;
  }
}

// ---- pipelined 32x64 tile (no guards): 8 WMMAs/k-step, B reused by 2 A fragments ----
__device__ __forceinline__ void gemm_32x64(const _Float16* __restrict__ A, int lda,
                                           const _Float16* __restrict__ Bt, int ldb,
                                           int m0, int n0, int K, v8f acc[8]) {
  v16h a0 = load_a16<false>(A, lda, m0,      0, 0);
  v16h a1 = load_a16<false>(A, lda, m0 + 16, 0, 0);
  v16h b0 = load_bt16<false>(Bt, ldb, 0, n0,      0);
  v16h b1 = load_bt16<false>(Bt, ldb, 0, n0 + 16, 0);
  v16h b2 = load_bt16<false>(Bt, ldb, 0, n0 + 32, 0);
  v16h b3 = load_bt16<false>(Bt, ldb, 0, n0 + 48, 0);
  for (int k0 = 0; k0 < K; k0 += 32) {
    int k1 = k0 + 32; if (k1 >= K) k1 = 0;
    v16h na0 = load_a16<false>(A, lda, m0,      k1, 0);
    v16h na1 = load_a16<false>(A, lda, m0 + 16, k1, 0);
    v16h nb0 = load_bt16<false>(Bt, ldb, k1, n0,      0);
    v16h nb1 = load_bt16<false>(Bt, ldb, k1, n0 + 16, 0);
    v16h nb2 = load_bt16<false>(Bt, ldb, k1, n0 + 32, 0);
    v16h nb3 = load_bt16<false>(Bt, ldb, k1, n0 + 48, 0);
    acc[0] = wmma_f16(a0, b0, acc[0]);
    acc[1] = wmma_f16(a0, b1, acc[1]);
    acc[2] = wmma_f16(a0, b2, acc[2]);
    acc[3] = wmma_f16(a0, b3, acc[3]);
    acc[4] = wmma_f16(a1, b0, acc[4]);
    acc[5] = wmma_f16(a1, b1, acc[5]);
    acc[6] = wmma_f16(a1, b2, acc[6]);
    acc[7] = wmma_f16(a1, b3, acc[7]);
    a0 = na0; a1 = na1; b0 = nb0; b1 = nb1; b2 = nb2; b3 = nb3;
  }
}

__device__ __forceinline__ float gelu_exact(float x) {
  return 0.5f * x * (1.0f + erff(x * 0.70710678118654752f));
}

// f32 store: EPI 0 = bias store, 2 = bias(+nullable) + residual accumulate
template<int EPI>
__device__ __forceinline__ void store_f32(float* C, int ldc, int m0, int n0,
                                          int Mlim, v8f acc, const float* bias) {
  const int lane = threadIdx.x & 31;
  const int n = n0 + (lane & 15);
  const int half = lane >> 4;
  const float bn = (bias != nullptr) ? bias[n] : 0.0f;
#pragma unroll
  for (int r = 0; r < 8; ++r) {
    const int m = m0 + r + (half << 3);
    if (m < Mlim) {
      float* p = C + (size_t)m * ldc + n;
      float v = acc[r] + bn;
      if (EPI == 2) v += *p;
      *p = v;
    }
  }
}

// f16 store: EPI 0 = bias, 1 = bias+gelu, 3 = scale (no bias)
template<int EPI>
__device__ __forceinline__ void store_f16(_Float16* C, int ldc, int m0, int n0,
                                          int Mlim, int Nlim, v8f acc,
                                          const float* bias, float scale) {
  const int lane = threadIdx.x & 31;
  const int n = n0 + (lane & 15);
  const int half = lane >> 4;
  if (n >= Nlim) return;
  const float bn = (EPI == 3) ? 0.0f : bias[n];
#pragma unroll
  for (int r = 0; r < 8; ++r) {
    const int m = m0 + r + (half << 3);
    if (m < Mlim) {
      float v = acc[r];
      if (EPI == 3) v *= scale; else v += bn;
      if (EPI == 1) v = gelu_exact(v);
      C[(size_t)m * ldc + n] = (_Float16)v;
    }
  }
}

// transposed f16 store (for v^T): Ct[n][m], contiguous v8h per lane
__device__ __forceinline__ void store_ct_f16(_Float16* Ct, int ldct, int m0, int n0,
                                             v8f acc, const float* bias) {
  const int lane = threadIdx.x & 31;
  const int n = n0 + (lane & 15);
  const int half = lane >> 4;
  const float bn = bias[n];
  v8h o;
#pragma unroll
  for (int r = 0; r < 8; ++r) o[r] = (_Float16)(acc[r] + bn);
  *(v8h*)(Ct + (size_t)n * ldct + m0 + (half << 3)) = o;
}

__device__ __forceinline__ int gwave() {
  return blockIdx.x * (blockDim.x >> 5) + (threadIdx.x >> 5);
}

// ---------------- prep / elementwise kernels ----------------
__global__ void k_patch_gather(const float* __restrict__ img,
                               _Float16* __restrict__ patches, int total) {
  int i = blockIdx.x * blockDim.x + threadIdx.x;
  if (i >= total) return;
  int kk = i % (C_ * PS_ * PS_);
  int rest = i / (C_ * PS_ * PS_);
  int p = rest % P_;
  int n = rest / P_;
  int c  = kk >> 8;
  int iy = (kk >> 4) & 15;
  int ix = kk & 15;
  int py = p / NP_;
  int px = p % NP_;
  size_t src = (((size_t)(n * C_ + c) * IMGW) + (py * PS_ + iy)) * IMGW + (px * PS_ + ix);
  patches[i] = (_Float16)img[src];
}

// transpose+convert weights: in (batch, K, N) f32 -> out (batch, N, K) f16
__global__ void k_cvt_transpose(const float* __restrict__ in, _Float16* __restrict__ out,
                                int K, int N, int total) {
  int i = blockIdx.x * blockDim.x + threadIdx.x;
  if (i >= total) return;
  int kn = K * N;
  int b = i / kn, r = i % kn;
  int k = r / N, n = r % N;
  out[(size_t)b * kn + (size_t)n * K + k] = (_Float16)in[i];
}

__global__ void k_cls_init(const float* __restrict__ cls, float* __restrict__ x) {
  int i = blockIdx.x * blockDim.x + threadIdx.x;
  if (i >= NB_ * D_) return;
  int n = i / D_, d = i % D_;
  x[(size_t)n * S_ * D_ + d] = cls[d];
}

__global__ void k_add_pos(float* __restrict__ x, const float* __restrict__ pos, int total) {
  int i = blockIdx.x * blockDim.x + threadIdx.x;
  if (i >= total) return;
  int d = i % D_;
  int s = (i / D_) % S_;
  x[i] += pos[s * D_ + d];
}

__global__ void k_zero_attpad(_Float16* att, int rows) {
  int i = blockIdx.x * blockDim.x + threadIdx.x;
  if (i >= rows * (ATTLD - S_)) return;
  int row = i / (ATTLD - S_), c = S_ + i % (ATTLD - S_);
  att[(size_t)row * ATTLD + c] = (_Float16)0.0f;
}
__global__ void k_zero_vtpad(_Float16* vt, int rows) {
  int i = blockIdx.x * blockDim.x + threadIdx.x;
  if (i >= rows * 16) return;
  int row = i / 16, c = 208 + i % 16;
  vt[(size_t)row * ATTLD + c] = (_Float16)0.0f;
}

__global__ void k_layernorm(const float* __restrict__ x, const float* __restrict__ w,
                            const float* __restrict__ b, _Float16* __restrict__ out) {
  int row = blockIdx.x;  // NB_*S_
  const float* xr = x + (size_t)row * D_;
  _Float16* orow = out + (size_t)row * D_;
  __shared__ float red[256];
  int tid = threadIdx.x;
  float s = 0.f;
  for (int d = tid; d < D_; d += 256) s += xr[d];
  red[tid] = s; __syncthreads();
  for (int o = 128; o > 0; o >>= 1) { if (tid < o) red[tid] += red[tid + o]; __syncthreads(); }
  float mu = red[0] / (float)D_;
  __syncthreads();
  float var = 0.f;
  for (int d = tid; d < D_; d += 256) { float t = xr[d] - mu; var += t * t; }
  red[tid] = var; __syncthreads();
  for (int o = 128; o > 0; o >>= 1) { if (tid < o) red[tid] += red[tid + o]; __syncthreads(); }
  float rstd = rsqrtf(red[0] / (float)D_ + 1e-5f);
  __syncthreads();
  for (int d = tid; d < D_; d += 256) orow[d] = (_Float16)((xr[d] - mu) * rstd * w[d] + b[d]);
}

// ---------------- WMMA GEMM kernels ----------------
#define EMB_MT 13
#define EMB_NG 12
__global__ void k_embed_gemm(const _Float16* __restrict__ Ah, const _Float16* __restrict__ Bt,
                             const float* __restrict__ bm, float* __restrict__ x) {
  int w = gwave();
  if (w >= NB_ * EMB_MT * EMB_NG) return;
  int ng = w % EMB_NG; int t = w / EMB_NG;
  int mt = t % EMB_MT; int n = t / EMB_MT;
  const _Float16* A = Ah + (size_t)n * P_ * D_;
  float* C = x + ((size_t)n * S_ + 1) * D_;
  int m0 = mt * 16, n0 = ng * 64;
  v8f acc[4] = {};
  gemm_16x64<true, false>(A, D_, P_, Bt, D_, 0, m0, n0, D_, acc);
#pragma unroll
  for (int j = 0; j < 4; ++j)
    store_f32<0>(C, D_, m0, n0 + j * 16, P_, acc[j], bm);
}

#define QKV_MT 13
__global__ void k_head_proj(const _Float16* __restrict__ hin, const _Float16* __restrict__ Wt,
                            const float* __restrict__ bias, _Float16* __restrict__ outq) {
  int w = gwave();
  if (w >= NB_ * H_ * QKV_MT) return;
  int mt = w % QKV_MT; int t = w / QKV_MT;
  int hd = t % H_; int n = t / H_;
  const _Float16* A = hin + (size_t)n * S_ * D_ + hd * DH_;
  const _Float16* B = Wt + (size_t)hd * DH_ * DH_;
  const float* bi = bias + hd * DH_;
  _Float16* C = outq + (size_t)n * S_ * D_ + hd * DH_;
  int m0 = mt * 16;
  v8f acc[4] = {};
  gemm_16x64<true, false>(A, D_, S_, B, DH_, 0, m0, 0, DH_, acc);
#pragma unroll
  for (int j = 0; j < 4; ++j)
    store_f16<0>(C, D_, m0, j * 16, S_, DH_, acc[j], bi, 1.0f);
}

__global__ void k_head_proj_v(const _Float16* __restrict__ hin, const _Float16* __restrict__ Wt,
                              const float* __restrict__ bias, _Float16* __restrict__ vt) {
  int w = gwave();
  if (w >= NB_ * H_ * QKV_MT) return;
  int mt = w % QKV_MT; int t = w / QKV_MT;
  int hd = t % H_; int n = t / H_;
  const _Float16* A = hin + (size_t)n * S_ * D_ + hd * DH_;
  const _Float16* B = Wt + (size_t)hd * DH_ * DH_;
  const float* bi = bias + hd * DH_;
  _Float16* Ct = vt + (size_t)(n * H_ + hd) * DH_ * ATTLD;
  int m0 = mt * 16;
  v8f acc[4] = {};
  gemm_16x64<true, false>(A, D_, S_, B, DH_, 0, m0, 0, DH_, acc);
#pragma unroll
  for (int j = 0; j < 4; ++j)
    store_ct_f16(Ct, ATTLD, m0, j * 16, acc[j], bi);
}

#define SC_MT 13
#define SC_NG 4
__global__ void k_scores(const _Float16* __restrict__ qh, const _Float16* __restrict__ kh,
                         _Float16* __restrict__ att, float scale) {
  int w = gwave();
  if (w >= NB_ * H_ * SC_MT * SC_NG) return;
  int ng = w % SC_NG; int t1 = w / SC_NG;
  int mt = t1 % SC_MT; int t2 = t1 / SC_MT;
  int hd = t2 % H_; int n = t2 / H_;
  const _Float16* A  = qh + (size_t)n * S_ * D_ + hd * DH_;
  const _Float16* Bt = kh + (size_t)n * S_ * D_ + hd * DH_;
  _Float16* C = att + (size_t)(n * H_ + hd) * S_ * ATTLD;
  int m0 = mt * 16, n0 = ng * 64;
  v8f acc[4] = {};
  gemm_16x64<true, true>(A, D_, S_, Bt, D_, S_, m0, n0, DH_, acc);
#pragma unroll
  for (int j = 0; j < 4; ++j)
    store_f16<3>(C, ATTLD, m0, n0 + j * 16, S_, S_, acc[j], nullptr, scale);
}

__global__ void k_softmax(_Float16* __restrict__ att, int rows) {
  int w = gwave();
  if (w >= rows) return;
  int lane = threadIdx.x & 31;
  _Float16* r = att + (size_t)w * ATTLD;
  float mx = -1e30f;
  for (int t = lane; t < S_; t += 32) mx = fmaxf(mx, (float)r[t]);
#pragma unroll
  for (int o = 16; o > 0; o >>= 1) mx = fmaxf(mx, __shfl_xor(mx, o, 32));
  float ev[7];
  int cnt = 0;
  float sm = 0.f;
  for (int t = lane; t < S_; t += 32) { float e = expf((float)r[t] - mx); ev[cnt++] = e; sm += e; }
#pragma unroll
  for (int o = 16; o > 0; o >>= 1) sm += __shfl_xor(sm, o, 32);
  float inv = 1.0f / sm;
  cnt = 0;
  for (int t = lane; t < S_; t += 32) r[t] = (_Float16)(ev[cnt++] * inv);
}

__global__ void k_attn_out(const _Float16* __restrict__ att, const _Float16* __restrict__ vt,
                           float* __restrict__ x) {
  int w = gwave();
  if (w >= NB_ * H_ * QKV_MT) return;
  int mt = w % QKV_MT; int t = w / QKV_MT;
  int hd = t % H_; int n = t / H_;
  const _Float16* A  = att + (size_t)(n * H_ + hd) * S_ * ATTLD;
  const _Float16* Bt = vt + (size_t)(n * H_ + hd) * DH_ * ATTLD;
  float* C = x + (size_t)n * S_ * D_ + hd * DH_;
  int m0 = mt * 16;
  v8f acc[4] = {};
  gemm_16x64<true, false>(A, ATTLD, S_, Bt, ATTLD, 0, m0, 0, ATTLD, acc);
#pragma unroll
  for (int j = 0; j < 4; ++j)
    store_f32<2>(C, D_, m0, j * 16, S_, acc[j], nullptr);
}

#define F1_MT ((NB_ * S_) / 32)  // 197 exact (32-row tiles)
#define F1_NG (FF_ / 64)
__global__ void k_ffn1(const _Float16* __restrict__ hin, const _Float16* __restrict__ W1t,
                       const float* __restrict__ b1, _Float16* __restrict__ mout) {
  int w = gwave();
  if (w >= F1_MT * F1_NG) return;
  int ng = w % F1_NG; int mt = w / F1_NG;
  int m0 = mt * 32, n0 = ng * 64;
  v8f acc[8] = {};
  gemm_32x64(hin, D_, W1t, D_, m0, n0, D_, acc);
#pragma unroll
  for (int j = 0; j < 4; ++j) {
    store_f16<1>(mout, FF_, m0,      n0 + j * 16, NB_ * S_, FF_, acc[j],     b1, 1.0f);
    store_f16<1>(mout, FF_, m0 + 16, n0 + j * 16, NB_ * S_, FF_, acc[4 + j], b1, 1.0f);
  }
}

#define F2_NG (D_ / 64)
__global__ void k_ffn2(const _Float16* __restrict__ mbuf, const _Float16* __restrict__ W2t,
                       const float* __restrict__ b2, float* __restrict__ x) {
  int w = gwave();
  if (w >= F1_MT * F2_NG) return;
  int ng = w % F2_NG; int mt = w / F2_NG;
  int m0 = mt * 32, n0 = ng * 64;
  v8f acc[8] = {};
  gemm_32x64(mbuf, FF_, W2t, FF_, m0, n0, FF_, acc);
#pragma unroll
  for (int j = 0; j < 4; ++j) {
    store_f32<2>(x, D_, m0,      n0 + j * 16, NB_ * S_, acc[j],     b2);
    store_f32<2>(x, D_, m0 + 16, n0 + j * 16, NB_ * S_, acc[4 + j], b2);
  }
}

// ---------------- classification head + softmax ----------------
__global__ void k_head(const float* __restrict__ x, const float* __restrict__ Wout,
                       const float* __restrict__ bout, float* __restrict__ out) {
  int n = blockIdx.x;
  const float* cls = x + (size_t)n * S_ * D_;
  __shared__ float logits[OUT_];
  __shared__ float red[256];
  int tid = threadIdx.x;
  for (int o = tid; o < OUT_; o += 256) {
    float s = bout[o];
    for (int d = 0; d < D_; ++d) s += cls[d] * Wout[(size_t)d * OUT_ + o];
    logits[o] = s;
  }
  __syncthreads();
  float mx = -1e30f;
  for (int o = tid; o < OUT_; o += 256) mx = fmaxf(mx, logits[o]);
  red[tid] = mx; __syncthreads();
  for (int o = 128; o > 0; o >>= 1) { if (tid < o) red[tid] = fmaxf(red[tid], red[tid + o]); __syncthreads(); }
  mx = red[0]; __syncthreads();
  float sm = 0.f;
  for (int o = tid; o < OUT_; o += 256) sm += expf(logits[o] - mx);
  red[tid] = sm; __syncthreads();
  for (int o = 128; o > 0; o >>= 1) { if (tid < o) red[tid] += red[tid + o]; __syncthreads(); }
  float inv = 1.0f / red[0];
  __syncthreads();
  for (int o = tid; o < OUT_; o += 256) out[(size_t)n * OUT_ + o] = expf(logits[o] - mx) * inv;
}

// ---------------- driver ----------------
extern "C" void kernel_launch(void* const* d_in, const int* in_sizes, int n_in,
                              void* d_out, int out_size, void* d_ws, size_t ws_size,
                              hipStream_t stream) {
  (void)in_sizes; (void)n_in; (void)out_size; (void)ws_size;
  const float* images = (const float*)d_in[0];
  const float* Wm   = (const float*)d_in[1];
  const float* bm   = (const float*)d_in[2];
  const float* cls  = (const float*)d_in[3];
  const float* pos  = (const float*)d_in[4];
  const float* ln1w = (const float*)d_in[5];
  const float* ln1b = (const float*)d_in[6];
  const float* Wq   = (const float*)d_in[7];
  const float* bq   = (const float*)d_in[8];
  const float* Wk   = (const float*)d_in[9];
  const float* bk   = (const float*)d_in[10];
  const float* Wv   = (const float*)d_in[11];
  const float* bv   = (const float*)d_in[12];
  const float* ln2w = (const float*)d_in[13];
  const float* ln2b = (const float*)d_in[14];
  const float* W1   = (const float*)d_in[15];
  const float* b1   = (const float*)d_in[16];
  const float* W2   = (const float*)d_in[17];
  const float* b2   = (const float*)d_in[18];
  const float* Wout = (const float*)d_in[19];
  const float* bout = (const float*)d_in[20];
  float* out = (float*)d_out;

  const size_t XSZ = (size_t)NB_ * S_ * D_;
  char* p = (char*)d_ws;
  auto alloc = [&](size_t bytes) { char* r = p; p += (bytes + 255) & ~(size_t)255; return r; };
  float*     x    = (float*)alloc(XSZ * 4);
  _Float16*  hh   = (_Float16*)alloc(XSZ * 2);
  _Float16*  qh   = (_Float16*)alloc(XSZ * 2);
  _Float16*  kh   = (_Float16*)alloc(XSZ * 2);
  _Float16*  vt   = (_Float16*)alloc((size_t)NB_ * H_ * DH_ * ATTLD * 2);
  _Float16*  atth = (_Float16*)alloc((size_t)NB_ * H_ * S_ * ATTLD * 2);
  _Float16*  big  = (_Float16*)alloc((size_t)(NB_ * S_) * FF_ * 2);  // patches / mlp hidden
  _Float16*  Wmt  = (_Float16*)alloc((size_t)D_ * D_ * 2);
  _Float16*  Wqt  = (_Float16*)alloc((size_t)H_ * DH_ * DH_ * 2);
  _Float16*  Wkt  = (_Float16*)alloc((size_t)H_ * DH_ * DH_ * 2);
  _Float16*  Wvt  = (_Float16*)alloc((size_t)H_ * DH_ * DH_ * 2);
  _Float16*  W1t  = (_Float16*)alloc((size_t)D_ * FF_ * 2);
  _Float16*  W2t  = (_Float16*)alloc((size_t)FF_ * D_ * 2);
  _Float16*  patches = big;
  _Float16*  mff     = big;

  const int TPB = 256;
  auto blocksW = [](long waves) { return (int)((waves + 7) / 8); };
  auto blocksE = [](long elems) { return (int)((elems + 255) / 256); };

  // one-time per call: pads + static weight conversion
  k_zero_attpad<<<blocksE((long)NB_ * H_ * S_ * (ATTLD - S_)), TPB, 0, stream>>>(atth, NB_ * H_ * S_);
  k_zero_vtpad<<<blocksE((long)NB_ * H_ * DH_ * 16), TPB, 0, stream>>>(vt, NB_ * H_ * DH_);
  k_cvt_transpose<<<blocksE((long)D_ * D_), TPB, 0, stream>>>(Wm, Wmt, D_, D_, D_ * D_);

  // patch embedding + cls + pos
  {
    long tot = (long)NB_ * P_ * D_;
    k_patch_gather<<<blocksE(tot), TPB, 0, stream>>>(images, patches, (int)tot);
    k_cls_init<<<blocksE((long)NB_ * D_), TPB, 0, stream>>>(cls, x);
    k_embed_gemm<<<blocksW((long)NB_ * EMB_MT * EMB_NG), TPB, 0, stream>>>(patches, Wmt, bm, x);
    k_add_pos<<<blocksE((long)XSZ), TPB, 0, stream>>>(x, pos, (int)XSZ);
  }

  const float scale = 0.125f;  // 1/sqrt(DH)
  for (int l = 0; l < L_; ++l) {
    const size_t woff = (size_t)l * H_ * DH_ * DH_;
    k_cvt_transpose<<<blocksE((long)H_ * DH_ * DH_), TPB, 0, stream>>>(Wq + woff, Wqt, DH_, DH_, H_ * DH_ * DH_);
    k_cvt_transpose<<<blocksE((long)H_ * DH_ * DH_), TPB, 0, stream>>>(Wk + woff, Wkt, DH_, DH_, H_ * DH_ * DH_);
    k_cvt_transpose<<<blocksE((long)H_ * DH_ * DH_), TPB, 0, stream>>>(Wv + woff, Wvt, DH_, DH_, H_ * DH_ * DH_);
    k_cvt_transpose<<<blocksE((long)D_ * FF_), TPB, 0, stream>>>(W1 + (size_t)l * D_ * FF_, W1t, D_, FF_, D_ * FF_);
    k_cvt_transpose<<<blocksE((long)FF_ * D_), TPB, 0, stream>>>(W2 + (size_t)l * FF_ * D_, W2t, FF_, D_, FF_ * D_);

    const long wqkv = (long)NB_ * H_ * QKV_MT;
    k_layernorm<<<NB_ * S_, TPB, 0, stream>>>(x, ln1w + l * D_, ln1b + l * D_, hh);
    k_head_proj<<<blocksW(wqkv), TPB, 0, stream>>>(hh, Wqt, bq + (size_t)l * H_ * DH_, qh);
    k_head_proj<<<blocksW(wqkv), TPB, 0, stream>>>(hh, Wkt, bk + (size_t)l * H_ * DH_, kh);
    k_head_proj_v<<<blocksW(wqkv), TPB, 0, stream>>>(hh, Wvt, bv + (size_t)l * H_ * DH_, vt);
    k_scores<<<blocksW((long)NB_ * H_ * SC_MT * SC_NG), TPB, 0, stream>>>(qh, kh, atth, scale);
    k_softmax<<<blocksW((long)NB_ * H_ * S_), TPB, 0, stream>>>(atth, NB_ * H_ * S_);
    k_attn_out<<<blocksW(wqkv), TPB, 0, stream>>>(atth, vt, x);
    k_layernorm<<<NB_ * S_, TPB, 0, stream>>>(x, ln2w + l * D_, ln2b + l * D_, hh);
    k_ffn1<<<blocksW((long)F1_MT * F1_NG), TPB, 0, stream>>>(hh, W1t, b1 + (size_t)l * FF_, mff);
    k_ffn2<<<blocksW((long)F1_MT * F2_NG), TPB, 0, stream>>>(mff, W2t, b2 + (size_t)l * D_, x);
  }

  k_head<<<NB_, TPB, 0, stream>>>(x, Wout, bout, out);
}
